// GBottleneck_60748017434629
// MI455X (gfx1250) — compile-verified, matrix operand order checked by reading, not run on hardware
//
#include <hip/hip_runtime.h>
#include <hip/hip_bf16.h>

// ---------------------------------------------------------------------------
// Pixel2Mesh GBottleneck on gfx1250: bf16 WMMA dual-GEMM (X@W and X@L fused,
// 6 N-tiles blocked per wave for A reuse, software-pipelined A prefetch),
// f32 accumulate, atomic-f32 SpMM scatter, residual scan in-place in d_out.
// ---------------------------------------------------------------------------

typedef __attribute__((ext_vector_type(16))) __bf16 v16bf;
typedef __attribute__((ext_vector_type(8)))  float  v8f;

#define NODES   100000
#define HID     192
#define MTILES  6250          // NODES / 16 (exact)

union BF16x16 {
    v16bf  v;
    __bf16 e[16];
};

union RawA {                  // raw fp32 staging for one 16x32 A slice (per lane)
    float4 v4[4];
    float  f[16];             // f[0..7] -> k = kb+0..7 ; f[8..15] -> k = kb+16..23
};

// ---------------------------------------------------------------------------
// Pack an fp32 [K x N] weight matrix into WMMA B-operand bf16 blobs.
// Blob layout: [ktile][ntile][lane 0..31][elem 0..15] bf16, where for lane l:
//   n = ntile*16 + (l & 15)
//   elem j -> k = ktile*32 + ((l>=16)?8:0) + ((j>=8)?16:0) + (j&7)
// (matches ISA 7.12.2 16-bit A/B striping for wave32). OOB k/n -> 0.
// ---------------------------------------------------------------------------
__global__ __launch_bounds__(256)
void pack_weight_kernel(const float* __restrict__ W, __bf16* __restrict__ P,
                        int K, int Ncols, int ktiles, int ntiles)
{
    long long idx = (long long)blockIdx.x * blockDim.x + threadIdx.x;
    long long total = (long long)ntiles * ktiles * 512;
    if (idx >= total) return;
    int e     = (int)(idx & 15);
    int lane  = (int)((idx >> 4) & 31);
    long long blk = idx >> 9;                  // kt*ntiles + ntile
    int ntile = (int)(blk % ntiles);
    int kt    = (int)(blk / ntiles);
    int n = ntile * 16 + (lane & 15);
    int k = kt * 32 + ((lane >> 4) << 3) + ((e >> 3) << 4) + (e & 7);
    float v = (k < K && n < Ncols) ? W[(size_t)k * Ncols + n] : 0.0f;
    P[idx] = (__bf16)v;
}

// ---------------------------------------------------------------------------
// Fused dual GEMM with N-tile blocking: Y = X@W ; Z = X@L + bias.
// One wave owns a 16-row M-tile and NT consecutive N-tiles (NT*2 v8f accs).
// Block = 256 threads = 8 waves; grid = (ceil(MTILES/8), ntilesTot/NT).
// A loaded once per k-tile and prefetched one k-tile ahead (hides the HBM
// stream); B blobs are L1-resident and loaded directly per WMMA pair.
// ALIGN16: X rows are 16B-aligned (K multiple of 4) -> aligned float4 loads;
// otherwise dword-aligned b128 via memcpy for full tiles, guarded scalar tail.
// ---------------------------------------------------------------------------
template<int NT, bool ALIGN16>
__global__ __launch_bounds__(256)
void gemm_dual_kernel(const float* __restrict__ X, int K, int ktiles,
                      const __bf16* __restrict__ Wp, const __bf16* __restrict__ Lp,
                      const float* __restrict__ bias,
                      float* __restrict__ Y, float* __restrict__ Z,
                      int ncols, int ntilesTot)
{
    const int lane  = threadIdx.x & 31;
    const int wave  = threadIdx.x >> 5;
    const int mtile = blockIdx.x * 8 + wave;
    if (mtile >= MTILES) return;               // wave-uniform: EXEC stays full
    const int nt0 = blockIdx.y * NT;           // first n-tile of this pass

    const int khalf = (lane >> 4) << 3;        // 0 for lanes 0-15, 8 for 16-31
    const float* xrow = X + ((size_t)mtile * 16 + (lane & 15)) * (size_t)K;

    v8f accW[NT];
    v8f accL[NT];
#pragma unroll
    for (int t = 0; t < NT; ++t) { accW[t] = (v8f){}; accL[t] = (v8f){}; }

    RawA ra;
    auto loadA = [&](int kt) {
        const int kb = kt * 32 + khalf;
        if (ALIGN16) {                         // rows 16B-aligned
            ra.v4[0] = *(const float4*)(xrow + kb);
            ra.v4[1] = *(const float4*)(xrow + kb + 4);
            ra.v4[2] = *(const float4*)(xrow + kb + 16);
            ra.v4[3] = *(const float4*)(xrow + kb + 20);
        } else if ((kt + 1) * 32 <= K) {       // full tile, dword-aligned b128
            __builtin_memcpy(&ra.v4[0], xrow + kb,      16);
            __builtin_memcpy(&ra.v4[1], xrow + kb + 4,  16);
            __builtin_memcpy(&ra.v4[2], xrow + kb + 16, 16);
            __builtin_memcpy(&ra.v4[3], xrow + kb + 20, 16);
        } else {                               // partial tail tile, guarded
#pragma unroll
            for (int j = 0; j < 8; ++j) {
                const int k0 = kb + j;
                const int k1 = kb + 16 + j;
                ra.f[j]     = (k0 < K) ? xrow[k0] : 0.0f;
                ra.f[j + 8] = (k1 < K) ? xrow[k1] : 0.0f;
            }
        }
    };

    loadA(0);
    for (int kt = 0; kt < ktiles; ++kt) {
        BF16x16 a;
#pragma unroll
        for (int j = 0; j < 16; ++j) a.e[j] = (__bf16)ra.f[j];

        if (kt + 1 < ktiles) loadA(kt + 1);    // prefetch next A behind WMMAs

        const __bf16* wp = Wp + ((size_t)kt * ntilesTot + nt0) * 512 + (size_t)lane * 16;
        const __bf16* lp = Lp + ((size_t)kt * ntilesTot + nt0) * 512 + (size_t)lane * 16;
#pragma unroll
        for (int t = 0; t < NT; ++t) {
            const v16bf bwv = *(const v16bf*)(wp + (size_t)t * 512);
            const v16bf blv = *(const v16bf*)(lp + (size_t)t * 512);
            accW[t] = __builtin_amdgcn_wmma_f32_16x16x32_bf16(
                false, a.v, false, bwv, (short)0, accW[t], false, false);
            accL[t] = __builtin_amdgcn_wmma_f32_16x16x32_bf16(
                false, a.v, false, blv, (short)0, accL[t], false, false);
        }
    }

    // D layout: vgpr r holds row m = mtile*16 + r + 8*(lane>>4), col = lane&15.
    const int mbase = mtile * 16 + ((lane >> 4) << 3);
#pragma unroll
    for (int t = 0; t < NT; ++t) {
        const int ncol = (nt0 + t) * 16 + (lane & 15);
        if (ncol < ncols) {
            const float bv = bias[ncol];
#pragma unroll
            for (int r = 0; r < 8; ++r) {
                const size_t row = (size_t)(mbase + r);
                Y[row * ncols + ncol] = accW[t][r];
                Z[row * ncols + ncol] = accL[t][r] + bv;
            }
        }
    }
}

// ---------------------------------------------------------------------------
// SpMM scatter: out[dst] += Y[src] * ew  (one thread per (edge, feature)).
// ---------------------------------------------------------------------------
__global__ __launch_bounds__(256)
void spmm_scatter_kernel(const float* __restrict__ Y, const int* __restrict__ src,
                         const int* __restrict__ dst, const float* __restrict__ ew,
                         float* __restrict__ out, int F, long long total)
{
    long long tid = (long long)blockIdx.x * blockDim.x + threadIdx.x;
    if (tid >= total) return;
    const int       f = (int)(tid % F);
    const long long e = tid / F;
    const int s = src[e];
    const int d = dst[e];
    const float v = Y[(size_t)s * F + f] * ew[e];
    unsafeAtomicAdd(out + (size_t)d * F + f, v);
}

// ---------------------------------------------------------------------------
// Residual combine: x = (x + h) * 0.5
// ---------------------------------------------------------------------------
__global__ __launch_bounds__(256)
void residual_kernel(float* __restrict__ x, const float* __restrict__ h, size_t n)
{
    size_t i = (size_t)blockIdx.x * blockDim.x + threadIdx.x;
    if (i < n) x[i] = (x[i] + h[i]) * 0.5f;
}

// ---------------------------------------------------------------------------
extern "C" void kernel_launch(void* const* d_in, const int* in_sizes, int n_in,
                              void* d_out, int out_size, void* d_ws, size_t ws_size,
                              hipStream_t stream)
{
    const float* X_in  = (const float*)d_in[0];
    const int*   esrc  = (const int*)d_in[1];
    const int*   edst  = (const int*)d_in[2];
    const float* ew    = (const float*)d_in[3];
    const float* w_in  = (const float*)d_in[4];
    const float* l_in  = (const float*)d_in[5];
    const float* b_in  = (const float*)d_in[6];
    const float* bw    = (const float*)d_in[7];
    const float* bl    = (const float*)d_in[8];
    const float* bb    = (const float*)d_in[9];
    const float* w_out = (const float*)d_in[10];
    const float* l_out = (const float*)d_in[11];
    const float* b_out = (const float*)d_in[12];

    const long long E      = in_sizes[1];
    const int       IN_DIM = in_sizes[4] / HID;            // 963
    const int       kt_in  = (IN_DIM + 31) / 32;           // 31

    float* xout = (float*)d_out;                           // [NODES*3]
    float* xcat = (float*)d_out + (size_t)NODES * 3;       // [NODES*HID], scan state

    // ---- workspace carve-up ----
    char* ws = (char*)d_ws;
    const size_t NF = (size_t)NODES * HID;
    float* Y  = (float*)ws;  ws += NF * sizeof(float);
    float* H1 = (float*)ws;  ws += NF * sizeof(float);
    float* H2 = (float*)ws;  ws += NF * sizeof(float);
    const size_t in_blob  = (size_t)12 * kt_in * 512;      // bf16 elems
    const size_t hid_blob = (size_t)12 * 6 * 512;
    const size_t out_blob = (size_t)1 * 6 * 512;
    __bf16* Pw_in  = (__bf16*)ws; ws += in_blob * 2;
    __bf16* Pl_in  = (__bf16*)ws; ws += in_blob * 2;
    __bf16* Pbw    = (__bf16*)ws; ws += 12 * hid_blob * 2; // 6 blocks x 2 convs
    __bf16* Pbl    = (__bf16*)ws; ws += 12 * hid_blob * 2;
    __bf16* Pw_out = (__bf16*)ws; ws += out_blob * 2;
    __bf16* Pl_out = (__bf16*)ws; ws += out_blob * 2;

    // ---- pack all weights to bf16 WMMA layout ----
    auto pack = [&](const float* W, __bf16* P, int K, int N, int kt, int nt) {
        long long tot = (long long)nt * kt * 512;
        pack_weight_kernel<<<(int)((tot + 255) / 256), 256, 0, stream>>>(W, P, K, N, kt, nt);
    };
    pack(w_in, Pw_in, IN_DIM, HID, kt_in, 12);
    pack(l_in, Pl_in, IN_DIM, HID, kt_in, 12);
    for (int b = 0; b < 6; ++b)
        for (int i = 0; i < 2; ++i) {
            const int m = b * 2 + i;
            pack(bw + (size_t)m * HID * HID, Pbw + (size_t)m * hid_blob, HID, HID, 6, 12);
            pack(bl + (size_t)m * HID * HID, Pbl + (size_t)m * hid_blob, HID, HID, 6, 12);
        }
    pack(w_out, Pw_out, HID, 3, 6, 1);
    pack(l_out, Pl_out, HID, 3, 6, 1);

    // ---- gconv = dual GEMM (6 n-tiles per wave) + atomic scatter ----
    const int GB = (MTILES + 7) / 8;                        // 782 blocks of 8 waves
    auto gconv = [&](const float* Xs, int K, int kt, const __bf16* Pw, const __bf16* Pl,
                     const float* bias, float* Zout, int ncols, int ntilesTot, bool a16) {
        if (ntilesTot == 12) {
            dim3 grid(GB, 2);                               // 2 passes of 6 n-tiles
            if (a16)
                gemm_dual_kernel<6, true ><<<grid, 256, 0, stream>>>(
                    Xs, K, kt, Pw, Pl, bias, Y, Zout, ncols, 12);
            else
                gemm_dual_kernel<6, false><<<grid, 256, 0, stream>>>(
                    Xs, K, kt, Pw, Pl, bias, Y, Zout, ncols, 12);
        } else {                                            // final layer: 1 n-tile
            dim3 grid(GB, 1);
            gemm_dual_kernel<1, true ><<<grid, 256, 0, stream>>>(
                Xs, K, kt, Pw, Pl, bias, Y, Zout, ncols, 1);
        }
        const long long total = E * ncols;
        spmm_scatter_kernel<<<(int)((total + 255) / 256), 256, 0, stream>>>(
            Y, esrc, edst, ew, Zout, ncols, total);
    };

    // input layer: x -> xcat region of d_out (scan state lives there)
    gconv(X_in, IN_DIM, kt_in, Pw_in, Pl_in, b_in, xcat, HID, 12, false);

    // 6 residual blocks
    for (int b = 0; b < 6; ++b) {
        gconv(xcat, HID, 6, Pbw + (size_t)(2 * b) * hid_blob,
              Pbl + (size_t)(2 * b) * hid_blob, bb + (size_t)(2 * b) * HID, H1, HID, 12, true);
        gconv(H1,   HID, 6, Pbw + (size_t)(2 * b + 1) * hid_blob,
              Pbl + (size_t)(2 * b + 1) * hid_blob, bb + (size_t)(2 * b + 1) * HID, H2, HID, 12, true);
        residual_kernel<<<(int)((NF + 255) / 256), 256, 0, stream>>>(xcat, H2, NF);
    }

    // output layer: x_out (3 cols) into d_out[0 .. 3*NODES)
    gconv(xcat, HID, 6, Pw_out, Pl_out, b_out, xout, 3, 1, true);
}
